// HierarchicalAttention_43825846288477
// MI455X (gfx1250) — matrix operand order repaired and checked by reading, not compiled, not run on hardware
//
#include <hip/hip_runtime.h>
#include <stdint.h>

typedef __attribute__((ext_vector_type(16))) __bf16 v16bf;
typedef __attribute__((ext_vector_type(8)))  float  v8f;

#define NTOK   32768            // 4 * 8192 tokens
#define DM     512
#define ATT_SCALE 0.125f        // 1/sqrt(64)

static __device__ __forceinline__ unsigned short f2bs(float f) {
  __bf16 h = (__bf16)f;
  return __builtin_bit_cast(unsigned short, h);
}
static __device__ __forceinline__ unsigned int pack2(unsigned short lo, unsigned short hi) {
  return (unsigned int)lo | ((unsigned int)hi << 16);
}
// LDS byte offset = low 32 bits of the generic shared address (flat-LDS rule)
static __device__ __forceinline__ unsigned int lds_off(const void* p) {
  return (unsigned int)(size_t)p;
}
// 16B global -> LDS async copy (ASYNCcnt); gfx1250 VGLOBAL async op
static __device__ __forceinline__ void async_b128(unsigned int lds, const void* gaddr) {
  asm volatile("global_load_async_to_lds_b128 %0, %1, off"
               :: "v"(lds), "v"(gaddr) : "memory");
}
static __device__ __forceinline__ void wait_async0() {
  asm volatile("s_wait_asynccnt 0" ::: "memory");
}

union FragB { v16bf v; unsigned int u[8]; };

// ---------------------------------------------------------------------------
// 128x128-tile GEMM: C[M,512] = A_bf16[M,512] @ W_f32[512,512] + bias
// A tiles: double-buffered async-to-LDS pipeline. W tiles: f32->bf16 on VALU
// (overlaps in-flight async copies). grid=(M/128,4), block=256 (8 waves).
// ---------------------------------------------------------------------------
template <bool WRITEF>
__global__ __launch_bounds__(256) void gemm512_bf16(
    const unsigned short* __restrict__ A,
    const float* __restrict__ W,
    const float* __restrict__ bias,
    unsigned short* __restrict__ outb,
    float* __restrict__ outf)
{
  __shared__ unsigned int As[2][128 * 16];   // 128 rows x 32 bf16, double buffered
  __shared__ unsigned int Bs[2][16 * 128];   // 16 k-pairs x 128 cols (bf16x2)

  const int tid  = threadIdx.x;
  const int lane = tid & 31;
  const int wid  = tid >> 5;
  const int hi   = lane >> 4;
  const int ln   = lane & 15;
  const int waveM = wid >> 2;                // 0..1
  const int waveN = wid & 3;                 // 0..3
  const int mBase = blockIdx.x * 128;
  const int nBase = blockIdx.y * 128;

  v8f vzero = {0.f,0.f,0.f,0.f,0.f,0.f,0.f,0.f};
  v8f acc[4][2];
  #pragma unroll
  for (int i = 0; i < 4; i++)
    #pragma unroll
    for (int j = 0; j < 2; j++) acc[i][j] = vzero;

  auto issueA = [&](int kk, int buf) {
    #pragma unroll
    for (int i = 0; i < 2; i++) {            // 512 x 16B per tile, 2 per thread
      int idx = tid * 2 + i;                 // m*4 + seg
      int m = idx >> 2, seg = idx & 3;
      async_b128(lds_off(&As[buf][m * 16 + seg * 4]),
                 A + (size_t)(mBase + m) * 512 + kk + seg * 8);
    }
  };

  issueA(0, 0);
  for (int step = 0; step < 16; step++) {
    const int kk = step * 32;
    const int p  = step & 1;

    #pragma unroll
    for (int i = 0; i < 8; i++) {            // stage W tile (overlaps async A)
      int idx = tid * 8 + i;                 // k2*128 + n
      int k2 = idx >> 7, n = idx & 127;
      float w0 = W[(size_t)(kk + 2 * k2)     * 512 + nBase + n];
      float w1 = W[(size_t)(kk + 2 * k2 + 1) * 512 + nBase + n];
      Bs[p][idx] = pack2(f2bs(w0), f2bs(w1));
      if (i == 0 && kk + 32 < 512)           // prefetch next W tile into caches
        __builtin_prefetch(&W[(size_t)(kk + 32 + 2 * k2) * 512 + nBase + n], 0, 1);
    }

    wait_async0();                           // A[p] resident in LDS
    __syncthreads();                         // all waves' B stores + asyncs done
    if (step < 15) issueA(kk + 32, 1 - p);   // prefetch next A tile into LDS

    FragB af[4], bfr[2];
    #pragma unroll
    for (int sm = 0; sm < 4; sm++) {
      int m = waveM * 64 + sm * 16 + ln;
      #pragma unroll
      for (int r = 0; r < 8; r++) {
        int k2 = ((r & 4) << 1) + hi * 4 + (r & 3);   // A-frag k pattern / 2
        af[sm].u[r] = As[p][m * 16 + k2];
      }
    }
    #pragma unroll
    for (int sn = 0; sn < 2; sn++) {
      int n = waveN * 32 + sn * 16 + ln;
      #pragma unroll
      for (int r = 0; r < 8; r++)
        bfr[sn].u[r] = Bs[p][(((r & 4) << 1) + hi * 4 + (r & 3)) * 128 + n]; // B-frag k pattern / 2
    }
    #pragma unroll
    for (int sm = 0; sm < 4; sm++)
      #pragma unroll
      for (int sn = 0; sn < 2; sn++)
        acc[sm][sn] = __builtin_amdgcn_wmma_f32_16x16x32_bf16(
            false, af[sm].v, false, bfr[sn].v, (short)0, acc[sm][sn], false, false);
  }

  #pragma unroll
  for (int sn = 0; sn < 2; sn++) {
    int col = nBase + waveN * 32 + sn * 16 + ln;
    float bv = bias[col];
    #pragma unroll
    for (int sm = 0; sm < 4; sm++) {
      #pragma unroll
      for (int r = 0; r < 8; r++) {
        int row = mBase + waveM * 64 + sm * 16 + hi * 8 + r;
        float v = acc[sm][sn][r] + bv;
        size_t o = (size_t)row * 512 + col;
        outb[o] = f2bs(v);
        if (WRITEF) outf[o] = v;
      }
    }
  }
}

// ---------------------------------------------------------------------------
// Block-local attention, one (b, blk, head, q-tile of 128) per workgroup.
// K staged [key][64] via async-to-LDS; V staged transposed [dim][512];
// streaming max-free softmax; per-wave P tile bounced through LDS.
// ---------------------------------------------------------------------------
__global__ __launch_bounds__(256) void attn_kernel(
    const unsigned short* __restrict__ Qb,
    const unsigned short* __restrict__ Kb,
    const unsigned short* __restrict__ Vb,
    unsigned short* __restrict__ attnb)
{
  extern __shared__ unsigned short smem[];
  unsigned short* Ksh = smem;                 // 512*64
  unsigned short* Vtr = smem + 512 * 64;      // 64*512
  unsigned short* Psh = smem + 2 * 512 * 64;  // 8 waves * 16*32

  const int bx  = blockIdx.x;
  const int qt  = bx & 3;
  const int h   = (bx >> 2) & 7;
  const int blk = (bx >> 5) & 15;
  const int b   = bx >> 9;
  const size_t tokbase = (size_t)b * 8192 + (size_t)blk * 512;

  const int tid  = threadIdx.x;
  const int lane = tid & 31;
  const int wid  = tid >> 5;
  const int hi   = lane >> 4;
  const int ln   = lane & 15;

  // ---- stage K (async copies) and V (transposed, manual scatter) ----
  #pragma unroll
  for (int r2 = 0; r2 < 2; r2++) {
    int key = tid * 2 + r2;
    const unsigned short* srcK = Kb + (tokbase + key) * 512 + h * 64;
    unsigned int dstK = lds_off(Ksh + key * 64);
    #pragma unroll
    for (int i = 0; i < 8; i++) async_b128(dstK + i * 16, srcK + i * 8);

    const uint4* srcV = (const uint4*)(Vb + (tokbase + key) * 512 + h * 64);
    #pragma unroll
    for (int i = 0; i < 8; i++) {
      uint4 v = srcV[i];
      int d0 = i * 8;
      Vtr[(d0 + 0) * 512 + key] = (unsigned short)(v.x & 0xffffu);
      Vtr[(d0 + 1) * 512 + key] = (unsigned short)(v.x >> 16);
      Vtr[(d0 + 2) * 512 + key] = (unsigned short)(v.y & 0xffffu);
      Vtr[(d0 + 3) * 512 + key] = (unsigned short)(v.y >> 16);
      Vtr[(d0 + 4) * 512 + key] = (unsigned short)(v.z & 0xffffu);
      Vtr[(d0 + 5) * 512 + key] = (unsigned short)(v.z >> 16);
      Vtr[(d0 + 6) * 512 + key] = (unsigned short)(v.w & 0xffffu);
      Vtr[(d0 + 7) * 512 + key] = (unsigned short)(v.w >> 16);
    }
  }
  wait_async0();
  __syncthreads();

  const int qBase = qt * 128 + wid * 16;
  const unsigned short* gQ = Qb + tokbase * 512 + h * 64;

  FragB aq[2];                                // Q 16x64 as two 16x32 A-frags
  #pragma unroll
  for (int dh = 0; dh < 2; dh++)
    #pragma unroll
    for (int r = 0; r < 8; r++) {
      int k = dh * 32 + ((r & 4) << 2) + hi * 8 + (r & 3) * 2;
      aq[dh].u[r] = *(const unsigned int*)(gQ + (size_t)(qBase + ln) * 512 + k);
    }

  v8f vzero = {0.f,0.f,0.f,0.f,0.f,0.f,0.f,0.f};
  v8f acc[4] = {vzero, vzero, vzero, vzero};
  float l[8] = {0.f,0.f,0.f,0.f,0.f,0.f,0.f,0.f};
  unsigned short* Pw = Psh + wid * 512;       // this wave's 16x32 P tile

  for (int kc = 0; kc < 16; kc++) {           // 32 keys per chunk
    int kb0 = kc * 32;
    #pragma unroll
    for (int t = 0; t < 2; t++) {
      int keyb = kb0 + t * 16;
      FragB bk0, bk1;
      #pragma unroll
      for (int r = 0; r < 8; r++) {
        int koff = ((r & 4) << 2) + hi * 8 + (r & 3) * 2;
        bk0.u[r] = *(const unsigned int*)(Ksh + (keyb + ln) * 64 + koff);
        bk1.u[r] = *(const unsigned int*)(Ksh + (keyb + ln) * 64 + 32 + koff);
      }
      v8f s = __builtin_amdgcn_wmma_f32_16x16x32_bf16(
          false, aq[0].v, false, bk0.v, (short)0, vzero, false, false);
      s = __builtin_amdgcn_wmma_f32_16x16x32_bf16(
          false, aq[1].v, false, bk1.v, (short)0, s, false, false);
      #pragma unroll
      for (int r = 0; r < 8; r++) {
        float p = __expf(s[r] * ATT_SCALE);
        l[r] += p;                            // partial row-sum (this lane's keys)
        Pw[(hi * 8 + r) * 32 + t * 16 + ln] = f2bs(p);
      }
    }
    asm volatile("s_wait_dscnt 0" ::: "memory");  // P stores -> P loads (same wave)

    FragB ap;                                 // P as 16x32 A-frag
    #pragma unroll
    for (int r = 0; r < 8; r++) {
      int k = ((r & 4) << 2) + hi * 8 + (r & 3) * 2;
      ap.u[r] = *(const unsigned int*)(Pw + ln * 32 + k);
    }
    #pragma unroll
    for (int ns = 0; ns < 4; ns++) {          // out(16x64) += P(16x32) @ V(32x64)
      FragB bv;
      #pragma unroll
      for (int r = 0; r < 8; r++)
        bv.u[r] = *(const unsigned int*)(Vtr + (ns * 16 + ln) * 512 + kb0 + ((r & 4) << 2) + hi * 8 + (r & 3) * 2);
      acc[ns] = __builtin_amdgcn_wmma_f32_16x16x32_bf16(
          false, ap.v, false, bv.v, (short)0, acc[ns], false, false);
    }
  }

  // reduce row-sums across the 16-lane halves, invert
  #pragma unroll
  for (int r = 0; r < 8; r++) {
    float v = l[r];
    v += __shfl_xor(v, 1, 32);
    v += __shfl_xor(v, 2, 32);
    v += __shfl_xor(v, 4, 32);
    v += __shfl_xor(v, 8, 32);
    l[r] = 1.0f / v;
  }
  #pragma unroll
  for (int ns = 0; ns < 4; ns++)
    #pragma unroll
    for (int r = 0; r < 8; r++) {
      int row = qBase + hi * 8 + r;
      int col = h * 64 + ns * 16 + ln;
      attnb[(tokbase + row) * 512 + col] = f2bs(acc[ns][r] * l[r]);
    }
}

// ---------------------------------------------------------------------------
// Gate GEMM: logits[M,512] = [local_bf16 | glob_bf16(block)] @ gate_w[1024,512]
// Same async double-buffered pipeline; epilogue: sigmoid blend in fp32.
// ---------------------------------------------------------------------------
__global__ __launch_bounds__(256) void gate_kernel(
    const unsigned short* __restrict__ localb,
    const unsigned short* __restrict__ globb,   // 64 x 512 bf16
    const float* __restrict__ gate_w,           // 1024 x 512
    const float* __restrict__ gate_b,
    const float* __restrict__ localf,
    const float* __restrict__ globf,            // 64 x 512 f32
    float* __restrict__ out)
{
  __shared__ unsigned int As[2][128 * 16];
  __shared__ unsigned int Bs[2][16 * 128];

  const int tid  = threadIdx.x;
  const int lane = tid & 31;
  const int wid  = tid >> 5;
  const int hi   = lane >> 4;
  const int ln   = lane & 15;
  const int waveM = wid >> 2, waveN = wid & 3;
  const int mBase = blockIdx.x * 128;
  const int nBase = blockIdx.y * 128;
  const int gb = mBase >> 9;                  // block row (b*16+blk), uniform in tile

  v8f vzero = {0.f,0.f,0.f,0.f,0.f,0.f,0.f,0.f};
  v8f acc[4][2];
  #pragma unroll
  for (int i = 0; i < 4; i++)
    #pragma unroll
    for (int j = 0; j < 2; j++) acc[i][j] = vzero;

  auto issueA = [&](int kk, int buf) {
    #pragma unroll
    for (int i = 0; i < 2; i++) {
      int idx = tid * 2 + i;
      int m = idx >> 2, seg = idx & 3;
      const unsigned short* g = (kk < 512)
          ? localb + (size_t)(mBase + m) * 512 + kk + seg * 8
          : globb + (size_t)gb * 512 + (kk - 512) + seg * 8;
      async_b128(lds_off(&As[buf][m * 16 + seg * 4]), g);
    }
  };

  issueA(0, 0);
  for (int step = 0; step < 32; step++) {
    const int kk = step * 32;
    const int p  = step & 1;

    #pragma unroll
    for (int i = 0; i < 8; i++) {
      int idx = tid * 8 + i;
      int k2 = idx >> 7, n = idx & 127;
      float w0 = gate_w[(size_t)(kk + 2 * k2)     * 512 + nBase + n];
      float w1 = gate_w[(size_t)(kk + 2 * k2 + 1) * 512 + nBase + n];
      Bs[p][idx] = pack2(f2bs(w0), f2bs(w1));
      if (i == 0 && kk + 32 < 1024)
        __builtin_prefetch(&gate_w[(size_t)(kk + 32 + 2 * k2) * 512 + nBase + n], 0, 1);
    }

    wait_async0();
    __syncthreads();
    if (step < 31) issueA(kk + 32, 1 - p);

    FragB af[4], bfr[2];
    #pragma unroll
    for (int sm = 0; sm < 4; sm++) {
      int m = waveM * 64 + sm * 16 + ln;
      #pragma unroll
      for (int r = 0; r < 8; r++)
        af[sm].u[r] = As[p][m * 16 + ((r & 4) << 1) + hi * 4 + (r & 3)];
    }
    #pragma unroll
    for (int sn = 0; sn < 2; sn++) {
      int n = waveN * 32 + sn * 16 + ln;
      #pragma unroll
      for (int r = 0; r < 8; r++)
        bfr[sn].u[r] = Bs[p][(((r & 4) << 1) + hi * 4 + (r & 3)) * 128 + n];
    }
    #pragma unroll
    for (int sm = 0; sm < 4; sm++)
      #pragma unroll
      for (int sn = 0; sn < 2; sn++)
        acc[sm][sn] = __builtin_amdgcn_wmma_f32_16x16x32_bf16(
            false, af[sm].v, false, bfr[sn].v, (short)0, acc[sm][sn], false, false);
  }

  #pragma unroll
  for (int sn = 0; sn < 2; sn++) {
    int col = nBase + waveN * 32 + sn * 16 + ln;
    float bv = gate_b[col];
    float gg = globf[(size_t)gb * 512 + col];
    #pragma unroll
    for (int sm = 0; sm < 4; sm++) {
      #pragma unroll
      for (int r = 0; r < 8; r++) {
        int row = mBase + waveM * 64 + sm * 16 + hi * 8 + r;
        size_t o = (size_t)row * 512 + col;
        float g = 1.0f / (1.0f + __expf(-(acc[sm][sn][r] + bv)));
        out[o] = g * localf[o] + (1.0f - g) * gg;
      }
    }
  }
}

// ---------------------------------------------------------------------------
// Tiny global path (scalar fp32 — negligible FLOPs)
// ---------------------------------------------------------------------------
__global__ void cvt_bf16_kernel(const float* __restrict__ in,
                                unsigned short* __restrict__ out, int n) {
  int i = blockIdx.x * blockDim.x + threadIdx.x;
  if (i < n) out[i] = f2bs(in[i]);
}

__global__ void pool_kernel(const float* __restrict__ x, float* __restrict__ pooled) {
  int idx = blockIdx.x * 256 + threadIdx.x;    // 64*512
  int row = idx >> 9, d = idx & 511;
  float s = 0.f;
  for (int t = 0; t < 512; t++) s += x[((size_t)(row * 512 + t)) * 512 + d];
  pooled[idx] = s * (1.0f / 512.0f);
}

__global__ void gproj_kernel(const float* __restrict__ pooled,
    const float* __restrict__ qw, const float* __restrict__ qb2,
    const float* __restrict__ kw, const float* __restrict__ kb2,
    const float* __restrict__ vw, const float* __restrict__ vb2,
    float* __restrict__ Qg, float* __restrict__ Kg, float* __restrict__ Vg) {
  int idx = blockIdx.x * 256 + threadIdx.x;    // 3 * 64 * 512
  int which = idx >> 15;
  int rem = idx & 32767;
  int row = rem >> 9, n = rem & 511;
  const float* W = (which == 0) ? qw : (which == 1) ? kw : vw;
  const float* B = (which == 0) ? qb2 : (which == 1) ? kb2 : vb2;
  float* O = (which == 0) ? Qg : (which == 1) ? Kg : Vg;
  float s = B[n];
  for (int k = 0; k < 512; k++) s += pooled[row * 512 + k] * W[(size_t)k * 512 + n];
  O[rem] = s;
}

__global__ void gattn_kernel(const float* __restrict__ Qg, const float* __restrict__ Kg,
                             const float* __restrict__ Vg, float* __restrict__ gout) {
  int idx = blockIdx.x * 256 + threadIdx.x;    // 4*16*8*64
  int d = idx & 63, h2 = (idx >> 6) & 7, q = (idx >> 9) & 15, b = idx >> 13;
  float sc[16];
  float mx = -1e30f;
  #pragma unroll
  for (int key = 0; key < 16; key++) {
    float s = 0.f;
    for (int dd = 0; dd < 64; dd++)
      s += Qg[(b * 16 + q) * 512 + h2 * 64 + dd] * Kg[(b * 16 + key) * 512 + h2 * 64 + dd];
    sc[key] = s * ATT_SCALE;
    mx = fmaxf(mx, sc[key]);
  }
  float denom = 0.f, num = 0.f;
  #pragma unroll
  for (int key = 0; key < 16; key++) {
    float e = __expf(sc[key] - mx);
    denom += e;
    num += e * Vg[(b * 16 + key) * 512 + h2 * 64 + d];
  }
  gout[(b * 16 + q) * 512 + h2 * 64 + d] = num / denom;
}

__global__ void gout_kernel(const float* __restrict__ gatt,
                            const float* __restrict__ gow, const float* __restrict__ gob,
                            float* __restrict__ globf, unsigned short* __restrict__ globb) {
  int idx = blockIdx.x * 256 + threadIdx.x;    // 64*512
  int row = idx >> 9, n = idx & 511;
  float s = gob[n];
  for (int k = 0; k < 512; k++) s += gatt[row * 512 + k] * gow[(size_t)k * 512 + n];
  globf[idx] = s;
  globb[idx] = f2bs(s);
}

// ---------------------------------------------------------------------------
extern "C" void kernel_launch(void* const* d_in, const int* in_sizes, int n_in,
                              void* d_out, int out_size, void* d_ws, size_t ws_size,
                              hipStream_t stream) {
  const float* x      = (const float*)d_in[0];
  const float* lq_w   = (const float*)d_in[1];  const float* lq_b   = (const float*)d_in[2];
  const float* lk_w   = (const float*)d_in[3];  const float* lk_b   = (const float*)d_in[4];
  const float* lv_w   = (const float*)d_in[5];  const float* lv_b   = (const float*)d_in[6];
  const float* lo_w   = (const float*)d_in[7];  const float* lo_b   = (const float*)d_in[8];
  const float* gq_w   = (const float*)d_in[9];  const float* gq_b   = (const float*)d_in[10];
  const float* gk_w   = (const float*)d_in[11]; const float* gk_b   = (const float*)d_in[12];
  const float* gv_w   = (const float*)d_in[13]; const float* gv_b   = (const float*)d_in[14];
  const float* go_w   = (const float*)d_in[15]; const float* go_b   = (const float*)d_in[16];
  const float* gate_w = (const float*)d_in[17]; const float* gate_b = (const float*)d_in[18];
  float* out = (float*)d_out;

  char* ws = (char*)d_ws;
  size_t off = 0;
  auto alloc = [&](size_t bytes) -> char* {
    char* p = ws + off;
    off += (bytes + 255) & ~(size_t)255;
    return p;
  };
  unsigned short* xb = (unsigned short*)alloc((size_t)NTOK * DM * 2);
  unsigned short* Qb = (unsigned short*)alloc((size_t)NTOK * DM * 2);
  unsigned short* Kb = (unsigned short*)alloc((size_t)NTOK * DM * 2);
  unsigned short* Vb = (unsigned short*)alloc((size_t)NTOK * DM * 2);
  float*          localf = (float*)alloc((size_t)NTOK * DM * 4);
  float* pooled = (float*)alloc(64 * 512 * 4);
  float* Qg     = (float*)alloc(64 * 512 * 4);
  float* Kg     = (float*)alloc(64 * 512 * 4);
  float* Vg     = (float*)alloc(64 * 512 * 4);
  float* gatt   = (float*)alloc(64 * 512 * 4);
  float* globf  = (float*)alloc(64 * 512 * 4);
  unsigned short* globb = (unsigned short*)alloc(64 * 512 * 2);
  unsigned short* attnb  = xb;   // reuse: xb dead after QKV GEMMs
  unsigned short* localb = Qb;   // reuse: Qb dead after attention

  dim3 gemmGrid(NTOK / 128, 4);

  cvt_bf16_kernel<<<(NTOK * DM) / 256, 256, 0, stream>>>(x, xb, NTOK * DM);

  gemm512_bf16<false><<<gemmGrid, 256, 0, stream>>>(xb, lq_w, lq_b, Qb, nullptr);
  gemm512_bf16<false><<<gemmGrid, 256, 0, stream>>>(xb, lk_w, lk_b, Kb, nullptr);
  gemm512_bf16<false><<<gemmGrid, 256, 0, stream>>>(xb, lv_w, lv_b, Vb, nullptr);

  size_t attn_lds = (size_t)(2 * 512 * 64 + 8 * 512) * sizeof(unsigned short);
  attn_kernel<<<2048, 256, attn_lds, stream>>>(Qb, Kb, Vb, attnb);

  gemm512_bf16<true><<<gemmGrid, 256, 0, stream>>>(attnb, lo_w, lo_b, localb, localf);

  pool_kernel<<<128, 256, 0, stream>>>(x, pooled);
  gproj_kernel<<<384, 256, 0, stream>>>(pooled, gq_w, gq_b, gk_w, gk_b, gv_w, gv_b, Qg, Kg, Vg);
  gattn_kernel<<<128, 256, 0, stream>>>(Qg, Kg, Vg, gatt);
  gout_kernel<<<128, 256, 0, stream>>>(gatt, go_w, go_b, globf, globb);

  gate_kernel<<<gemmGrid, 256, 0, stream>>>(localb, globb, gate_w, gate_b, localf, globf, out);
}
